// BaichuanAttentionOriginal_16793322127632
// MI455X (gfx1250) — compile-verified, matrix operand order checked by reading, not run on hardware
//
#include <hip/hip_runtime.h>
#include <hip/hip_bf16.h>
#include <math.h>

// ---------------------------------------------------------------------------
// Baichuan attention for MI455X (gfx1250, wave32, WMMA bf16 16x16x32).
// B=2, S=1024, H=5120, NH=40, D=128.
// ---------------------------------------------------------------------------

typedef __attribute__((ext_vector_type(16))) __bf16 v16bf;
typedef __attribute__((ext_vector_type(8)))  float  v8f;
typedef __attribute__((ext_vector_type(4)))  int    v4i;

union FragBF {            // 16 bf16 = 32 bytes, loaded as two 16B chunks
    v16bf v;
    uint4 q[2];
};

__device__ __forceinline__ unsigned short f2bf(float f) {
    // round-to-nearest-even truncation to bf16
    unsigned int u = __float_as_uint(f);
    unsigned int r = u + 0x7FFFu + ((u >> 16) & 1u);
    return (unsigned short)(r >> 16);
}

__device__ __forceinline__ unsigned pack_bf2(float a, float b) {
#if __has_builtin(__builtin_amdgcn_cvt_pk_bf16_f32)
    auto r = __builtin_amdgcn_cvt_pk_bf16_f32(a, b);   // v_cvt_pk_bf16_f32
    unsigned u;
    __builtin_memcpy(&u, &r, 4);
    return u;
#else
    return (unsigned)f2bf(a) | ((unsigned)f2bf(b) << 16);
#endif
}

__device__ __forceinline__ uint2 f4_to_bf4(float4 f) {
    uint2 r;
    r.x = pack_bf2(f.x, f.y);
    r.y = pack_bf2(f.z, f.w);
    return r;
}

// ---- CDNA5 async global->LDS copy (ASYNCcnt path), guarded ----------------
#if __has_builtin(__builtin_amdgcn_global_load_async_to_lds_b128)
#define HAVE_ASYNC_LDS 1
// builtin expects addrspace-qualified v4i pointers (AS1 src, AS3 dst)
typedef __attribute__((address_space(1))) v4i gv4i;
typedef __attribute__((address_space(3))) v4i lv4i;
#else
#define HAVE_ASYNC_LDS 0
#endif

__device__ __forceinline__ void copy16_g2lds(unsigned short* lds_dst,
                                             const unsigned short* gsrc) {
#if HAVE_ASYNC_LDS
    __builtin_amdgcn_global_load_async_to_lds_b128(
        (gv4i*)gsrc, (lv4i*)lds_dst, 0, 0);
#else
    *(uint4*)lds_dst = *(const uint4*)gsrc;
#endif
}

__device__ __forceinline__ void wait_async_all() {
#if HAVE_ASYNC_LDS
#if __has_builtin(__builtin_amdgcn_s_wait_asynccnt)
    __builtin_amdgcn_s_wait_asynccnt(0);
#else
    asm volatile("s_wait_asynccnt 0" ::: "memory");
#endif
#endif
}

// ---------------------------------------------------------------------------
// Tiled GEMM:  C[M,N] = A[M,K] (fp32, row-major) * W[N,K]^T (fp32, row-major)
// Block tile 128x128, K-step 32, 8 waves (2 M x 4 N), per-wave 4x2 WMMA tiles.
// mode 0: scatter QKV -> Q (scaled, bf16 [b,h,s,d]), K (bf16 [b,h,s,d]),
//         V (bf16 [b,h,d,s]  == pre-transposed for the PV WMMA B-operand)
// mode 1: plain fp32 store to out[M,N]
// ---------------------------------------------------------------------------
#define BM 128
#define BN 128
#define BKK 32

__global__ __launch_bounds__(256)
void gemm_bf16_kernel(const float* __restrict__ A, const float* __restrict__ W,
                      int M, int N, int K, int mode,
                      unsigned short* __restrict__ outQ,
                      unsigned short* __restrict__ outK,
                      unsigned short* __restrict__ outVt,
                      float* __restrict__ outF)
{
    __shared__ __attribute__((aligned(16))) unsigned short lA[BM * BKK];
    __shared__ __attribute__((aligned(16))) unsigned short lB[BN * BKK];

    const int tid  = threadIdx.x;
    const int lane = tid & 31;
    const int wid  = tid >> 5;
    const int wm   = wid & 1;    // 0..1  (64 rows each)
    const int wn   = wid >> 1;   // 0..3  (32 cols each)
    const int m0   = blockIdx.x * BM;   // M fastest -> L2 reuse of W strip
    const int n0   = blockIdx.y * BN;

    v8f acc[4][2];
#pragma unroll
    for (int i = 0; i < 4; ++i)
#pragma unroll
        for (int j = 0; j < 2; ++j)
#pragma unroll
            for (int r = 0; r < 8; ++r) acc[i][j][r] = 0.0f;

    const int row = tid >> 1;            // 0..127
    const int seg = (tid & 1) * 16;      // k-offset 0 / 16
    const int fr  = lane & 15;           // fragment row (M or N within tile)
    const int kb  = (lane < 16) ? 0 : 8; // ISA 16-bit operand K-swizzle

    for (int k0 = 0; k0 < K; k0 += BKK) {
        __syncthreads();
        // fp32 global -> bf16 LDS (each thread: 16 A floats + 16 W floats)
        const float4* pa = (const float4*)(A + (size_t)(m0 + row) * K + k0 + seg);
        const float4* pb = (const float4*)(W + (size_t)(n0 + row) * K + k0 + seg);
#pragma unroll
        for (int i = 0; i < 4; ++i) {
            *(uint2*)&lA[row * BKK + seg + i * 4] = f4_to_bf4(pa[i]);
            *(uint2*)&lB[row * BKK + seg + i * 4] = f4_to_bf4(pb[i]);
        }
        if (k0 + BKK < K) {   // prefetch next k-tile -> global_prefetch_b8
            __builtin_prefetch(A + (size_t)(m0 + row) * K + k0 + BKK + seg, 0, 1);
            __builtin_prefetch(W + (size_t)(n0 + row) * K + k0 + BKK + seg, 0, 1);
        }
        __syncthreads();

        FragBF af[4], bf[2];
#pragma unroll
        for (int im = 0; im < 4; ++im) {
            const unsigned short* p = &lA[(wm * 64 + im * 16 + fr) * BKK + kb];
            af[im].q[0] = *(const uint4*)(p);
            af[im].q[1] = *(const uint4*)(p + 16);
        }
#pragma unroll
        for (int in = 0; in < 2; ++in) {
            const unsigned short* p = &lB[(wn * 32 + in * 16 + fr) * BKK + kb];
            bf[in].q[0] = *(const uint4*)(p);
            bf[in].q[1] = *(const uint4*)(p + 16);
        }
#pragma unroll
        for (int im = 0; im < 4; ++im)
#pragma unroll
            for (int in = 0; in < 2; ++in)
                acc[im][in] = __builtin_amdgcn_wmma_f32_16x16x32_bf16(
                    false, af[im].v, false, bf[in].v,
                    (short)0, acc[im][in], false, false);
    }

    // epilogue (C layout: VGPR r, lane l -> row r+8*(l>=16), col l&15)
    const int hi    = lane >> 4;
    const int col_l = lane & 15;
    const float qscale = 0.08838834764831845f;   // 1/sqrt(128)
#pragma unroll
    for (int im = 0; im < 4; ++im) {
#pragma unroll
        for (int in = 0; in < 2; ++in) {
#pragma unroll
            for (int r = 0; r < 8; ++r) {
                const int m = m0 + wm * 64 + im * 16 + r + 8 * hi;
                const int n = n0 + wn * 32 + in * 16 + col_l;
                const float v = acc[im][in][r];
                if (mode == 0) {
                    const int t   = n / 5120;
                    const int rem = n - t * 5120;
                    const int h   = rem >> 7;
                    const int d   = rem & 127;
                    const int b   = m >> 10;
                    const int s   = m & 1023;
                    if (t == 0)
                        outQ[(((size_t)b * 40 + h) * 1024 + s) * 128 + d] = f2bf(v * qscale);
                    else if (t == 1)
                        outK[(((size_t)b * 40 + h) * 1024 + s) * 128 + d] = f2bf(v);
                    else
                        outVt[(((size_t)b * 40 + h) * 128 + d) * 1024 + s] = f2bf(v);
                } else {
                    outF[(size_t)m * N + n] = v;
                }
            }
        }
    }
}

// ---------------------------------------------------------------------------
// Flash attention with analytic ALiBi bias.
// Grid: (S/64, NH, B).  Block: 128 threads = 4 waves, each wave owns 16 q rows.
// ---------------------------------------------------------------------------
struct Slopes { float v[40]; };

__global__ __launch_bounds__(128)
void attn_kernel(const unsigned short* __restrict__ Qb,   // bf16 [b,h,s,d] (pre-scaled)
                 const unsigned short* __restrict__ Kb,   // bf16 [b,h,s,d]
                 const unsigned short* __restrict__ Vt,   // bf16 [b,h,d,s]
                 float* __restrict__ attnOut,             // fp32 [b,s,h*d]
                 Slopes sl)
{
    __shared__ __attribute__((aligned(16))) unsigned short lK[64 * 128];  // [key][d]
    __shared__ __attribute__((aligned(16))) unsigned short lV[128 * 64];  // [d][key]
    __shared__ __attribute__((aligned(16))) unsigned short lP[4 * 16 * 64];

    const int tid  = threadIdx.x;
    const int lane = tid & 31;
    const int wid  = tid >> 5;
    const int qt   = blockIdx.x;
    const int h    = blockIdx.y;
    const int b    = blockIdx.z;
    const int q0   = qt * 64 + wid * 16;

    const size_t head = ((size_t)b * 40 + h) * 1024 * 128;
    const unsigned short* Qh = Qb + head;
    const unsigned short* Kh = Kb + head;
    const unsigned short* Vh = Vt + head;   // [d][s]

    const int fr = lane & 15;
    const int kb = (lane < 16) ? 0 : 8;
    const int hi = lane >> 4;
    const float slope = sl.v[h];

    // Q fragments kept in registers: 4 chunks of K=32 over D=128
    FragBF qf[4];
#pragma unroll
    for (int c = 0; c < 4; ++c) {
        const unsigned short* p = Qh + (size_t)(q0 + fr) * 128 + c * 32 + kb;
        qf[c].q[0] = *(const uint4*)p;
        qf[c].q[1] = *(const uint4*)(p + 16);
    }

    float mrow[8], lrow[8];
    v8f acc[8];
#pragma unroll
    for (int r = 0; r < 8; ++r) { mrow[r] = -INFINITY; lrow[r] = 0.0f; }
#pragma unroll
    for (int jd = 0; jd < 8; ++jd)
#pragma unroll
        for (int r = 0; r < 8; ++r) acc[jd][r] = 0.0f;

    unsigned short* myP = &lP[wid * 16 * 64];

    for (int kt = 0; kt <= qt; ++kt) {
        const int k0 = kt * 64;
        __syncthreads();
        {   // K tile: 64x128 bf16 (1024 x 16B / 128 threads) -> async DMA to LDS
            const unsigned short* ks = Kh + (size_t)k0 * 128;
#pragma unroll
            for (int i = 0; i < 8; ++i) {
                const int idx = (tid + i * 128) * 8;     // 16B chunk offset
                copy16_g2lds(lK + idx, ks + idx);
            }
            // V^T tile: each thread copies one d-row of 64 keys (8 x 16B)
            const unsigned short* vs = Vh + (size_t)tid * 1024 + k0;
#pragma unroll
            for (int i = 0; i < 8; ++i)
                copy16_g2lds(lV + tid * 64 + i * 8, vs + i * 8);
        }
        wait_async_all();   // s_wait_asynccnt 0 (when async path available)
        __syncthreads();

        // scores = Q * K^T  (4 key-subtiles x 4 d-chunks of WMMA)
        v8f sc[4];
#pragma unroll
        for (int jn = 0; jn < 4; ++jn) {
#pragma unroll
            for (int r = 0; r < 8; ++r) sc[jn][r] = 0.0f;
#pragma unroll
            for (int c = 0; c < 4; ++c) {
                const unsigned short* p = &lK[(jn * 16 + fr) * 128 + c * 32 + kb];
                FragBF kf;
                kf.q[0] = *(const uint4*)p;
                kf.q[1] = *(const uint4*)(p + 16);
                sc[jn] = __builtin_amdgcn_wmma_f32_16x16x32_bf16(
                    false, qf[c].v, false, kf.v, (short)0, sc[jn], false, false);
            }
        }

        // ALiBi bias + causal mask + online softmax
#pragma unroll
        for (int r = 0; r < 8; ++r) {
            const int q = q0 + r + 8 * hi;
            float s[4];
            float tmax = -INFINITY;
#pragma unroll
            for (int jn = 0; jn < 4; ++jn) {
                const int key = k0 + jn * 16 + fr;
                const float bias = (key <= q) ? slope * (float)key : -INFINITY;
                s[jn] = sc[jn][r] + bias;
                tmax = fmaxf(tmax, s[jn]);
            }
#pragma unroll
            for (int xm = 8; xm >= 1; xm >>= 1)
                tmax = fmaxf(tmax, __shfl_xor(tmax, xm, 32));
            const float mnew  = fmaxf(mrow[r], tmax);
            const float alpha = __expf(mrow[r] - mnew);
            float psum = 0.0f;
#pragma unroll
            for (int jn = 0; jn < 4; ++jn) {
                const float p = __expf(s[jn] - mnew);
                psum += p;
                myP[(r + 8 * hi) * 64 + jn * 16 + fr] = f2bf(p);  // C->A relayout via LDS
            }
#pragma unroll
            for (int xm = 8; xm >= 1; xm >>= 1)
                psum += __shfl_xor(psum, xm, 32);
            lrow[r] = lrow[r] * alpha + psum;
            mrow[r] = mnew;
#pragma unroll
            for (int jd = 0; jd < 8; ++jd) acc[jd][r] *= alpha;
        }

        // ensure P stores landed (LDS in-order per wave; block compiler motion)
        asm volatile("s_wait_dscnt 0" ::: "memory");

        FragBF pf[2];
#pragma unroll
        for (int c = 0; c < 2; ++c) {
            const unsigned short* p = &myP[fr * 64 + c * 32 + kb];
            pf[c].q[0] = *(const uint4*)p;
            pf[c].q[1] = *(const uint4*)(p + 16);
        }
        // O += P * V   (8 d-subtiles x 2 key-chunks)
#pragma unroll
        for (int jd = 0; jd < 8; ++jd) {
#pragma unroll
            for (int c = 0; c < 2; ++c) {
                const unsigned short* p = &lV[(jd * 16 + fr) * 64 + c * 32 + kb];
                FragBF vf;
                vf.q[0] = *(const uint4*)p;
                vf.q[1] = *(const uint4*)(p + 16);
                acc[jd] = __builtin_amdgcn_wmma_f32_16x16x32_bf16(
                    false, pf[c].v, false, vf.v, (short)0, acc[jd], false, false);
            }
        }
    }

    // normalize and store fp32 [b, q, h*128 + d]
#pragma unroll
    for (int r = 0; r < 8; ++r) {
        const float inv = (lrow[r] > 0.0f) ? (1.0f / lrow[r]) : 0.0f;
        const int q = q0 + r + 8 * hi;
        float* dst = attnOut + ((size_t)b * 1024 + q) * 5120 + h * 128;
#pragma unroll
        for (int jd = 0; jd < 8; ++jd)
            dst[jd * 16 + fr] = acc[jd][r] * inv;
    }
}

// ---------------------------------------------------------------------------
// Host side
// ---------------------------------------------------------------------------
static void fill_slopes(float* s, int n) {   // matches _get_interleave(n)
    int closest = 1;
    while (closest * 2 <= n) closest *= 2;
    const float start  = powf(2.0f, -powf(2.0f, -(log2f((float)closest) - 3.0f)));
    for (int i = 0; i < closest; ++i) s[i] = powf(start, (float)(i + 1));
    const float start2 = powf(2.0f, -powf(2.0f, -(log2f((float)(2 * closest)) - 3.0f)));
    for (int j = 0; j < n - closest; ++j) s[closest + j] = powf(start2, (float)(2 * j + 1));
}

extern "C" void kernel_launch(void* const* d_in, const int* in_sizes, int n_in,
                              void* d_out, int out_size, void* d_ws, size_t ws_size,
                              hipStream_t stream) {
    const float* hidden = (const float*)d_in[0];   // [2,1024,5120]
    const float* Wpack  = (const float*)d_in[1];   // [15360,5120]
    const float* Wo     = (const float*)d_in[2];   // [5120,5120]
    // d_in[3] (attention_mask) unused: ALiBi bias is computed analytically.

    const int Bb = 2, S = 1024, H = 5120, NH = 40, D = 128;
    const int M = Bb * S;               // 2048
    const size_t headElems = (size_t)Bb * NH * S * D;   // 10,485,760

    // workspace layout (≈100 MB): Q,K,V^T bf16 + attn fp32
    unsigned short* Qb = (unsigned short*)d_ws;
    unsigned short* Kb = Qb + headElems;
    unsigned short* Vt = Kb + headElems;
    float* attnBuf     = (float*)(Vt + headElems);

    Slopes sl;
    fill_slopes(sl.v, NH);

    // 1) packed QKV projection
    dim3 g0(M / 128, (3 * H) / 128);
    gemm_bf16_kernel<<<g0, dim3(256), 0, stream>>>(
        hidden, Wpack, M, 3 * H, H, 0, Qb, Kb, Vt, nullptr);

    // 2) ALiBi causal flash attention
    dim3 ga(S / 64, NH, Bb);
    attn_kernel<<<ga, dim3(128), 0, stream>>>(Qb, Kb, Vt, attnBuf, sl);

    // 3) output projection -> fp32 d_out
    dim3 g1(M / 128, H / 128);
    gemm_bf16_kernel<<<g1, dim3(256), 0, stream>>>(
        attnBuf, Wo, M, H, H, 1, nullptr, nullptr, nullptr, (float*)d_out);
}
